// GAT_50946902065159
// MI455X (gfx1250) — matrix operand order repaired and checked by reading, not compiled, not run on hardware
//
#include <hip/hip_runtime.h>
#include <hip/hip_bf16.h>

typedef __bf16 v16bf __attribute__((ext_vector_type(16)));
typedef float  v8f   __attribute__((ext_vector_type(8)));

#define NN    4096
#define FIN   512
#define NK    8
#define FO    128
#define KF    1024   // NK*FO
#define NEG_SLOPE 0.2f
#define TSTRIDE 40   // padded LDS row stride (bf16 elems): 20 dwords -> conflict-free b128

union V16B { uint4 q[2]; v16bf v; __bf16 h[16]; };
union V8B  { uint4 q;    __bf16 h[8]; };

// ---------------- workspace layout (bytes) ----------------
#define XB_OFF   ((size_t)0)                       // 4096*512 bf16 = 4 MB
#define WB_OFF   ((size_t)4194304)                 // 1024*512 bf16 = 1 MB
#define HT_OFF   ((size_t)5242880)                 // Ht[o][m] 1024*4096 bf16 = 8 MB
#define LEFT_OFF ((size_t)13631488)                // 8*4096 f32
#define RGHT_OFF ((size_t)13762560)
#define E1_OFF   ((size_t)13893632)
#define E2_OFF   ((size_t)14024704)
#define F1_OFF   ((size_t)14155776)
#define F2_OFF   ((size_t)14286848)                // end ~14.4 MB

// ---------------- fp32 -> bf16 convert ----------------
__global__ void cvt_bf16(const float* __restrict__ src, __bf16* __restrict__ dst, int n) {
    int i = blockIdx.x * blockDim.x + threadIdx.x;
    if (i < n) dst[i] = (__bf16)src[i];
}

// ---------------- projection GEMM: h = x @ W^T + b, stored transposed bf16 ----------------
// grid (16, 32), block 256 (8 waves). wave -> one 16-row n-tile, 64 o-columns.
__global__ void __launch_bounds__(256)
gemm_proj(const __bf16* __restrict__ xb, const __bf16* __restrict__ wb,
          const float* __restrict__ bias, __bf16* __restrict__ Ht) {
    const int lane = threadIdx.x & 31;
    const int wave = threadIdx.x >> 5;
    const int half = lane >> 4;
    const int rc   = lane & 15;
    const int n0   = (blockIdx.y * 8 + wave) * 16;
    const int o0   = blockIdx.x * 64;

    v8f acc[4] = {};
    const __bf16* arow = xb + (size_t)(n0 + rc) * FIN;

    for (int kk = 0; kk < FIN; kk += 32) {
        // A frag: row n0+rc; elems 0..7 = cols kk+8*half.. ; elems 8..15 = cols kk+16+8*half..
        V16B a;
        a.q[0] = *(const uint4*)(arow + kk + 8 * half);
        a.q[1] = *(const uint4*)(arow + kk + 16 + 8 * half);
#pragma unroll
        for (int t = 0; t < 4; ++t) {
            const int o = o0 + t * 16 + rc;
            const __bf16* wp = wb + (size_t)o * FIN + kk + 8 * half;
            V16B bfr;
            bfr.q[0] = *(const uint4*)(wp);
            bfr.q[1] = *(const uint4*)(wp + 16);
            acc[t] = __builtin_amdgcn_wmma_f32_16x16x32_bf16(
                false, a.v, false, bfr.v, (short)0, acc[t], false, false);
        }
    }
    // epilogue: + bias, write Ht[o][n] (D rows v+8*half are contiguous in Ht)
#pragma unroll
    for (int t = 0; t < 4; ++t) {
        const int o = o0 + t * 16 + rc;
        const float bo = bias[o];
        V8B pk;
#pragma unroll
        for (int v = 0; v < 8; ++v) pk.h[v] = (__bf16)(acc[t][v] + bo);
        *(uint4*)(Ht + (size_t)o * NN + n0 + 8 * half) = pk.q;
    }
}

// ---------------- left/right attention dots ----------------
// grid (16, 8), block 256: thread -> one (k, n)
__global__ void __launch_bounds__(256)
leftright(const __bf16* __restrict__ Ht, const float* __restrict__ a_left,
          const float* __restrict__ a_right, float* __restrict__ left,
          float* __restrict__ right) {
    const int n = blockIdx.x * blockDim.x + threadIdx.x;
    const int k = blockIdx.y;
    float al = 0.f, ar = 0.f;
#pragma unroll 4
    for (int f = 0; f < FO; ++f) {
        const float hv = (float)Ht[(size_t)(k * FO + f) * NN + n];
        al += hv * a_left[k * FO + f];
        ar += hv * a_right[k * FO + f];
    }
    left[k * NN + n]  = al;
    right[k * NN + n] = ar;
}

// ---------------- E1 = exp(right), E2 = exp(0.2 right) ----------------
__global__ void expfactors(const float* __restrict__ right, float* __restrict__ E1,
                           float* __restrict__ E2) {
    int i = blockIdx.x * blockDim.x + threadIdx.x;
    if (i < NK * NN) {
        const float r = right[i];
        E1[i] = __expf(r);
        E2[i] = __expf(NEG_SLOPE * r);
    }
}

// ---------------- masked row max -> F1 = exp(left-maxe), F2 = exp(0.2 left-maxe) ----------------
__global__ void __launch_bounds__(128)
rowmax(const unsigned char* __restrict__ mask, const float* __restrict__ right,
       const float* __restrict__ left, float* __restrict__ F1, float* __restrict__ F2) {
    __shared__ float sh[NK][128];
    const int n = blockIdx.x;
    const int t = threadIdx.x;
    float mx[NK];
#pragma unroll
    for (int k = 0; k < NK; ++k) mx[k] = -1e30f;
    const unsigned char* mrow = mask + (size_t)n * NN;
    for (int m = t; m < NN; m += 128) {
        if (mrow[m]) {
#pragma unroll
            for (int k = 0; k < NK; ++k) mx[k] = fmaxf(mx[k], right[k * NN + m]);
        }
    }
#pragma unroll
    for (int k = 0; k < NK; ++k) sh[k][t] = mx[k];
    __syncthreads();
    if (t < NK) {
        float mm = sh[t][0];
        for (int j = 1; j < 128; ++j) mm = fmaxf(mm, sh[t][j]);
        const float le = left[t * NN + n];
        const float s  = le + mm;
        const float maxe = s >= 0.f ? s : NEG_SLOPE * s;   // monotone leaky_relu
        F1[t * NN + n] = __expf(le - maxe);
        F2[t * NN + n] = __expf(NEG_SLOPE * le - maxe);
    }
}

// ---------------- fused softmax + aggregation + ELU ----------------
// grid (32, 8): y = head, x*8+wave = 16-row n-tile. Each wave: 16 rows x 128 cols.
// Ht B-tiles (shared by all 8 waves) staged via async global->LDS, double-buffered.
__global__ void __launch_bounds__(256)
gat_attn(const unsigned char* __restrict__ mask, const __bf16* __restrict__ Ht,
         const float* __restrict__ left, const float* __restrict__ right,
         const float* __restrict__ E1, const float* __restrict__ E2,
         const float* __restrict__ F1, const float* __restrict__ F2,
         float* __restrict__ out) {
    __shared__ __align__(16) __bf16 tile[2][FO * TSTRIDE];   // 2 x 10,240 B

    const int lane = threadIdx.x & 31;
    const int wave = threadIdx.x >> 5;
    const int half = lane >> 4;
    const int rc   = lane & 15;
    const int k    = blockIdx.y;
    const int n0   = (blockIdx.x * 8 + wave) * 16;
    const int nrow = n0 + rc;                 // A-matrix row handled by this lane

    const float thr = -left[k * NN + nrow];
    const float f1  = F1[k * NN + nrow];
    const float f2  = F2[k * NN + nrow];
    const float* Rk  = right + k * NN;
    const float* E1k = E1 + k * NN;
    const float* E2k = E2 + k * NN;
    const unsigned char* mrow = mask + (size_t)nrow * NN;
    const __bf16* Hk = Ht + (size_t)k * FO * NN;

    // async-stage one 128(f) x 32(m) bf16 tile: 512 16B chunks, 2 per thread
    const int c0   = threadIdx.x;            // chunk ids c0, c0+256
    const int row0 = c0 >> 2, part0 = c0 & 3;
    const int c1   = c0 + 256;
    const int row1 = c1 >> 2, part1 = c1 & 3;

#define STAGE(m0_, buf_)                                                          \
    {                                                                             \
        const __bf16* g0 = Hk + (size_t)row0 * NN + (m0_) + part0 * 8;            \
        const __bf16* g1 = Hk + (size_t)row1 * NN + (m0_) + part1 * 8;            \
        unsigned l0 = (unsigned)(size_t)&tile[buf_][row0 * TSTRIDE + part0 * 8];  \
        unsigned l1 = (unsigned)(size_t)&tile[buf_][row1 * TSTRIDE + part1 * 8];  \
        asm volatile("global_load_async_to_lds_b128 %0, %1, off"                  \
                     :: "v"(l0), "v"(g0) : "memory");                             \
        asm volatile("global_load_async_to_lds_b128 %0, %1, off"                  \
                     :: "v"(l1), "v"(g1) : "memory");                             \
    }

    STAGE(0, 0)

    v8f acc[8] = {};
    float ssum = 0.f;

    for (int m0 = 0; m0 < NN; m0 += 32) {
        const int buf = (m0 >> 5) & 1;
        // ---- build A fragment (P tile, 16x32 bf16) in registers ----
        V16B a;
#pragma unroll
        for (int g = 0; g < 2; ++g) {         // elems g*8..g*8+7 <-> K = g*16 + 8*half + j
            const int mb = m0 + g * 16 + 8 * half;
            const float4 r0  = *(const float4*)(Rk + mb);
            const float4 r1  = *(const float4*)(Rk + mb + 4);
            const float4 e1a = *(const float4*)(E1k + mb);
            const float4 e1b = *(const float4*)(E1k + mb + 4);
            const float4 e2a = *(const float4*)(E2k + mb);
            const float4 e2b = *(const float4*)(E2k + mb + 4);
            const uint2 mk   = *(const uint2*)(mrow + mb);
            const float rv[8]  = {r0.x, r0.y, r0.z, r0.w, r1.x, r1.y, r1.z, r1.w};
            const float e1v[8] = {e1a.x, e1a.y, e1a.z, e1a.w, e1b.x, e1b.y, e1b.z, e1b.w};
            const float e2v[8] = {e2a.x, e2a.y, e2a.z, e2a.w, e2b.x, e2b.y, e2b.z, e2b.w};
#pragma unroll
            for (int j = 0; j < 8; ++j) {
                const unsigned mw = (j < 4) ? mk.x : mk.y;
                const bool in  = ((mw >> (8 * (j & 3))) & 0xffu) != 0u;
                const bool pos = rv[j] >= thr;
                const float p  = in ? (pos ? f1 * e1v[j] : f2 * e2v[j]) : 0.f;
                ssum += p;
                a.h[g * 8 + j] = (__bf16)p;
            }
        }
        // ---- tile m0 staged & previous buffer free: sync, then prefetch next ----
        asm volatile("s_wait_asynccnt 0x0" ::: "memory");
        __syncthreads();
        if (m0 + 32 < NN) STAGE(m0 + 32, buf ^ 1)
        // ---- 8 f-tiles: B frags from LDS, WMMA accumulate ----
#pragma unroll
        for (int ft = 0; ft < 8; ++ft) {
            const __bf16* bp = &tile[buf][(ft * 16 + rc) * TSTRIDE + 8 * half];
            V16B bfr;
            bfr.q[0] = *(const uint4*)(bp);
            bfr.q[1] = *(const uint4*)(bp + 16);
            acc[ft] = __builtin_amdgcn_wmma_f32_16x16x32_bf16(
                false, a.v, false, bfr.v, (short)0, acc[ft], false, false);
        }
    }

    // row sums: lanes r and r+16 cover complementary K halves of row r
    const float T = ssum + __shfl_xor(ssum, 16, 32);
    const int base = 8 * half;                 // D rows for this lane: base..base+7
    float S[8];
#pragma unroll
    for (int v = 0; v < 8; ++v) S[v] = __shfl(T, base + v, 32);

#pragma unroll
    for (int ft = 0; ft < 8; ++ft) {
        const int f = k * FO + ft * 16 + rc;
#pragma unroll
        for (int v = 0; v < 8; ++v) {
            const float val = acc[ft][v] / S[v];
            out[(size_t)(n0 + base + v) * KF + f] = val > 0.f ? val : expm1f(val);
        }
    }
#undef STAGE
}

extern "C" void kernel_launch(void* const* d_in, const int* in_sizes, int n_in,
                              void* d_out, int out_size, void* d_ws, size_t ws_size,
                              hipStream_t stream) {
    const float*         x       = (const float*)d_in[0];
    const unsigned char* mask    = (const unsigned char*)d_in[1];   // jnp.bool_: 1 byte
    const float*         W       = (const float*)d_in[2];
    const float*         b       = (const float*)d_in[3];
    const float*         a_left  = (const float*)d_in[4];
    const float*         a_right = (const float*)d_in[5];
    float*               out     = (float*)d_out;

    char* ws = (char*)d_ws;
    __bf16* xb  = (__bf16*)(ws + XB_OFF);
    __bf16* wb  = (__bf16*)(ws + WB_OFF);
    __bf16* Ht  = (__bf16*)(ws + HT_OFF);
    float* left = (float*)(ws + LEFT_OFF);
    float* rght = (float*)(ws + RGHT_OFF);
    float* E1   = (float*)(ws + E1_OFF);
    float* E2   = (float*)(ws + E2_OFF);
    float* F1   = (float*)(ws + F1_OFF);
    float* F2   = (float*)(ws + F2_OFF);

    cvt_bf16<<<(NN * FIN + 255) / 256, 256, 0, stream>>>(x, xb, NN * FIN);
    cvt_bf16<<<(KF * FIN + 255) / 256, 256, 0, stream>>>(W, wb, KF * FIN);
    gemm_proj<<<dim3(16, 32), 256, 0, stream>>>(xb, wb, b, Ht);
    leftright<<<dim3(16, NK), 256, 0, stream>>>(Ht, a_left, a_right, left, rght);
    expfactors<<<(NK * NN + 255) / 256, 256, 0, stream>>>(rght, E1, E2);
    rowmax<<<NN, 128, 0, stream>>>(mask, rght, left, F1, F2);
    gat_attn<<<dim3(32, NK), 256, 0, stream>>>(mask, Ht, left, rght, E1, E2, F1, F2, out);
}
